// CTC_CRF_LOSS_17583596110272
// MI455X (gfx1250) — matrix operand order repaired and checked by reading, not compiled, not run on hardware
//
#include <hip/hip_runtime.h>
#include <hip/hip_bf16.h>
#include <math.h>

typedef __attribute__((ext_vector_type(16))) __bf16 v16bf;
typedef __attribute__((ext_vector_type(8)))  __bf16 v8bf;
typedef __attribute__((ext_vector_type(8)))  float  v8f;

// Problem constants (N, T, V, L) = (64, 1024, 512, 64)
constexpr int kN = 64;
constexpr int kT = 1024;
constexpr int kV = 512;
constexpr int kL = 64;
constexpr int kS = 2 * kL + 1;           // 129 CTC states
constexpr float kNEG = -1e30f;

// LDS padded strides (dwords / bf16 elems) to avoid bank conflicts
constexpr int kAS = 516;                  // f32 row stride for state vector
constexpr int kPS = 520;                  // bf16 row stride for p = exp(a - amax)

__device__ __forceinline__ float lae(float x, float y) {
    float mx = fmaxf(x, y);
    float mn = fminf(x, y);
    return (mx <= -1e29f) ? kNEG : mx + log1pf(__expf(mn - mx));
}

// ------------------------------------------------------------------
// Et[n][k] = exp(trans[k][n])  (transposed, bf16) — B-matrix source
// ------------------------------------------------------------------
__global__ void prep_et_kernel(const float* __restrict__ trans,
                               __bf16* __restrict__ Et) {
    int idx = blockIdx.x * blockDim.x + threadIdx.x;   // 0 .. V*V-1
    int n = idx >> 9;
    int k = idx & (kV - 1);
    Et[idx] = (__bf16)__expf(trans[k * kV + n]);
}

// ------------------------------------------------------------------
// lse[n,t] = logsumexp_v logits[n,t,v]   (one wave per row)
// ------------------------------------------------------------------
__global__ void lse_kernel(const float* __restrict__ logits,
                           float* __restrict__ lse) {
    int lane = threadIdx.x & 31;
    int row  = blockIdx.x * 8 + (threadIdx.x >> 5);    // 8 rows / 256-thread block
    const float* p = logits + (size_t)row * kV;
    float mx = -INFINITY;
    #pragma unroll
    for (int i = 0; i < 16; ++i) mx = fmaxf(mx, p[lane + i * 32]);
    #pragma unroll
    for (int off = 16; off > 0; off >>= 1) mx = fmaxf(mx, __shfl_xor(mx, off, 32));
    float s = 0.f;
    #pragma unroll
    for (int i = 0; i < 16; ++i) s += __expf(p[lane + i * 32] - mx);
    #pragma unroll
    for (int off = 16; off > 0; off >>= 1) s += __shfl_xor(s, off, 32);
    if (lane == 0) lse[row] = mx + __logf(s);
}

// ------------------------------------------------------------------
// CTC forward NLL: one workgroup per sequence, S=129 states
// ------------------------------------------------------------------
__global__ void ctc_kernel(const float* __restrict__ logits,
                           const int* __restrict__ labels,
                           const float* __restrict__ lse,
                           float* __restrict__ nll) {
    __shared__ int   ext[kS];
    __shared__ float abuf[2][kS];
    const int n   = blockIdx.x;
    const int tid = threadIdx.x;

    if (tid < kS) ext[tid] = (tid & 1) ? labels[n * kL + (tid >> 1)] : 0;
    __syncthreads();

    bool allow = false;
    if (tid < kS && tid >= 2) allow = (ext[tid] != 0) && (ext[tid] != ext[tid - 2]);

    const float* lgn  = logits + (size_t)n * kT * kV;
    const float* lsen = lse + n * kT;

    if (tid < kS) {
        float a0;
        if (tid == 0)      a0 = lgn[ext[0]] - lsen[0];
        else if (tid == 1) a0 = lgn[ext[1]] - lsen[0];
        else               a0 = kNEG;
        abuf[0][tid] = a0;
    }
    __syncthreads();

    int cur = 0;
    for (int t = 1; t < kT; ++t) {
        if (tid < kS) {
            float a0 = abuf[cur][tid];
            float a1 = (tid >= 1) ? abuf[cur][tid - 1] : kNEG;
            float a2 = allow ? abuf[cur][tid - 2] : kNEG;
            float lp = lgn[(size_t)t * kV + ext[tid]] - lsen[t];
            abuf[cur ^ 1][tid] = lae(lae(a0, a1), a2) + lp;
        }
        __syncthreads();
        cur ^= 1;
    }
    if (tid == 0) nll[n] = -lae(abuf[cur][kS - 1], abuf[cur][kS - 2]);
}

// ------------------------------------------------------------------
// Dense-graph forward (logZ): 16 sequences per WG, 8 waves.
// Per step: a_new = amax + log( exp(a-amax) @ exp(trans) ) + logits[t]
// GEMM via v_wmma_f32_16x16x32_bf16; M=16 seqs, N=K=512 states.
// ------------------------------------------------------------------
__global__ void __launch_bounds__(256)
den_forward_kernel(const float* __restrict__ logits,
                   const __bf16* __restrict__ Et,
                   float* __restrict__ logZ) {
    __shared__ float  Abuf[16 * kAS];      // f32 state vectors
    __shared__ __bf16 Pbuf[16 * kPS];      // bf16 exp(a - amax), WMMA A source
    __shared__ float  s_part[16][16];
    __shared__ float  s_amax[16];

    const int tid  = threadIdx.x;          // 0..255
    const int seq0 = blockIdx.x * 16;
    const int wave = tid >> 5;             // 0..7 -> owns N columns [wave*64, wave*64+64)
    const int lane = tid & 31;
    const int hi   = lane >> 4;            // lane half (selects K sub-chunk)
    const int lq   = lane & 15;            // row (A: M) / column (B,D: N) within tile
    const int mrow = tid >> 4;             // 0..15 for reductions
    const int chnk = tid & 15;

    // init: a0 = logits[:, 0, :]
    for (int idx = tid; idx < 16 * kV; idx += 256) {
        int m = idx >> 9, v = idx & (kV - 1);
        Abuf[m * kAS + v] = logits[(size_t)(seq0 + m) * kT * kV + v];
    }
    __syncthreads();

    for (int t = 1; t < kT; ++t) {
        // ---- phase 1a: per-sequence max ----
        float mx = -INFINITY;
        #pragma unroll 8
        for (int i = 0; i < 32; ++i) mx = fmaxf(mx, Abuf[mrow * kAS + chnk * 32 + i]);
        s_part[mrow][chnk] = mx;
        __syncthreads();
        if (tid < 16) {
            float mm = -INFINITY;
            #pragma unroll
            for (int j = 0; j < 16; ++j) mm = fmaxf(mm, s_part[tid][j]);
            s_amax[tid] = mm;
        }
        __syncthreads();
        // ---- phase 1b: p = exp(a - amax) -> bf16 ----
        {
            float am = s_amax[mrow];
            #pragma unroll 8
            for (int i = 0; i < 32; ++i) {
                int v = chnk * 32 + i;
                Pbuf[mrow * kPS + v] = (__bf16)__expf(Abuf[mrow * kAS + v] - am);
            }
        }
        __syncthreads();

        // ---- phase 2: b = p @ E, 4 N-tiles x 16 K-tiles of 16x16x32 bf16 WMMA ----
        v8f acc[4] = {};
        #pragma unroll 4
        for (int kt = 0; kt < 16; ++kt) {
            const int kbase = kt * 32;
            // A (16x32 bf16): lanes 0-15 -> M=lane, K=kbase+{0..7,16..23};
            //                 lanes16-31 -> M=lane-16, K=kbase+{8..15,24..31}
            const __bf16* pr = &Pbuf[lq * kPS + kbase + hi * 8];
            v8bf alo = *(const v8bf*)(pr);
            v8bf ahi = *(const v8bf*)(pr + 16);
            v16bf Amat = __builtin_shufflevector(alo, ahi,
                0, 1, 2, 3, 4, 5, 6, 7, 8, 9, 10, 11, 12, 13, 14, 15);
            #pragma unroll
            for (int nt = 0; nt < 4; ++nt) {
                const int nbase = (wave * 4 + nt) * 16;
                // B (32x16 bf16): lane holds column N=lq, K=kbase+hi*8+{0..7,16..23}
                const __bf16* br = &Et[(size_t)(nbase + lq) * kV + kbase + hi * 8];
                v8bf blo = *(const v8bf*)(br);
                v8bf bhi = *(const v8bf*)(br + 16);
                v16bf Bmat = __builtin_shufflevector(blo, bhi,
                    0, 1, 2, 3, 4, 5, 6, 7, 8, 9, 10, 11, 12, 13, 14, 15);
                acc[nt] = __builtin_amdgcn_wmma_f32_16x16x32_bf16(
                    false, Amat, false, Bmat, (short)0, acc[nt], false, false);
            }
        }

        // ---- phase 3: a_new = amax + log(b) + logits[t] ----
        #pragma unroll
        for (int nt = 0; nt < 4; ++nt) {
            const int n = (wave * 4 + nt) * 16 + lq;
            #pragma unroll
            for (int r = 0; r < 8; ++r) {
                const int m = r + hi * 8;   // C/D layout: VGPR r, lane-half -> row
                float anew = s_amax[m] + __logf(acc[nt][r])
                           + logits[(size_t)(seq0 + m) * kT * kV + (size_t)t * kV + n];
                Abuf[m * kAS + n] = anew;
            }
        }
        __syncthreads();
    }

    // ---- final: logZ = logsumexp over states ----
    float mx = -INFINITY;
    for (int i = 0; i < 32; ++i) mx = fmaxf(mx, Abuf[mrow * kAS + chnk * 32 + i]);
    s_part[mrow][chnk] = mx;
    __syncthreads();
    if (tid < 16) {
        float mm = -INFINITY;
        for (int j = 0; j < 16; ++j) mm = fmaxf(mm, s_part[tid][j]);
        s_amax[tid] = mm;
    }
    __syncthreads();
    {
        float am = s_amax[mrow];
        float s = 0.f;
        for (int i = 0; i < 32; ++i) s += __expf(Abuf[mrow * kAS + chnk * 32 + i] - am);
        s_part[mrow][chnk] = s;
    }
    __syncthreads();
    if (tid < 16) {
        float tot = 0.f;
        for (int j = 0; j < 16; ++j) tot += s_part[tid][j];
        logZ[seq0 + tid] = s_amax[tid] + __logf(tot);
    }
}

// ------------------------------------------------------------------
// out = sum_n(logZ[n] - 1.1 * nll[n]) / N
// ------------------------------------------------------------------
__global__ void finalize_kernel(const float* __restrict__ logZ,
                                const float* __restrict__ nll,
                                float* __restrict__ out) {
    __shared__ float buf[kN];
    int tid = threadIdx.x;
    buf[tid] = logZ[tid] - 1.1f * nll[tid];
    __syncthreads();
    if (tid == 0) {
        float s = 0.f;
        for (int i = 0; i < kN; ++i) s += buf[i];
        out[0] = s / (float)kN;
    }
}

extern "C" void kernel_launch(void* const* d_in, const int* in_sizes, int n_in,
                              void* d_out, int out_size, void* d_ws, size_t ws_size,
                              hipStream_t stream) {
    const float* logits = (const float*)d_in[0];
    const int*   labels = (const int*)d_in[1];
    // d_in[2]=lx, d_in[3]=ly: full lengths, unused by the reference math
    const float* trans  = (const float*)d_in[4];
    float* out = (float*)d_out;

    char* ws = (char*)d_ws;
    __bf16* Et  = (__bf16*)ws;                                   // 512 KB
    float* lse  = (float*)(ws + 512 * 1024);                     // 256 KB
    float* nll  = (float*)(ws + 512 * 1024 + 256 * 1024);        // 256 B
    float* logZ = (float*)(ws + 512 * 1024 + 256 * 1024 + 256);  // 256 B

    prep_et_kernel<<<(kV * kV) / 256, 256, 0, stream>>>(trans, Et);
    lse_kernel<<<(kN * kT) / 8, 256, 0, stream>>>(logits, lse);
    ctc_kernel<<<kN, 160, 0, stream>>>(logits, labels, lse, nll);
    den_forward_kernel<<<kN / 16, 256, 0, stream>>>(logits, Et, logZ);
    finalize_kernel<<<1, kN, 0, stream>>>(logZ, nll, out);
}
